// MutaelEncoder_19894288515584
// MI455X (gfx1250) — matrix operand order repaired and checked by reading, not compiled, not run on hardware
//
#include <hip/hip_runtime.h>

// ---------------------------------------------------------------------------
// MutaelEncoder for MI455X (gfx1250): 4 layers of dual-graph SAGEConv.
// 16x [2048^3] GEMMs on bf16 WMMA (v_wmma_f32_16x16x32_bf16).
// All GEMM operands are pre-converted (and weights pre-transposed) to bf16
// [row][k] row-major once per launch, so GEMM tile staging is a pure DMA:
// double-buffered TENSOR_LOAD_TO_LDS (TDM) with s_wait_tensorcnt, falling
// back to b128 copy staging when the TDM builtin is unavailable.
// Workspace requirement: ~224 MB.
// ---------------------------------------------------------------------------

#define NDIM   2048
#define LAYERS 4
#define EKNN   30720
#define EPPI   65536
#define NEGSL  0.01f
#define LDT    40            // padded K-stride (ushort units): 64B row + 16B pad
#define TILE_US (128 * LDT)  // ushorts per LDS tile region (5120)

typedef __attribute__((ext_vector_type(16))) __bf16 v16bf;
typedef __attribute__((ext_vector_type(8)))  float  v8f;

union FragBF { uint4 q[2]; v16bf v; };

__device__ __forceinline__ unsigned short f32_bf16(float f) {
    union { float f; unsigned u; } v; v.f = f;
    unsigned u = v.u;
    u += 0x7fffu + ((u >> 16) & 1u);   // round to nearest even
    return (unsigned short)(u >> 16);
}

__device__ __forceinline__ uint4 lds_q(const unsigned short* S, int idx_ushort) {
    return *(const uint4*)((const char*)S + (size_t)idx_ushort * 2);
}

// ---------------------------------------------------------------------------
// Tensor Data Mover tile load: 128 rows x 32 k (bf16) from a 2048x2048 bf16
// matrix into LDS [row][k] with LDT padding (pad 4 DWORDs every 16 DWORDs).
// ---------------------------------------------------------------------------
#if __has_builtin(__builtin_amdgcn_tensor_load_to_lds)
#define HAVE_TDM 1
typedef unsigned int v4u  __attribute__((ext_vector_type(4)));
typedef int          v8i_ __attribute__((ext_vector_type(8)));
typedef int          v4i_ __attribute__((ext_vector_type(4)));

__device__ __forceinline__ void tdm_load_tile(const unsigned short* __restrict__ G,
                                              int rbase, int kbase, unsigned ldsoff) {
    unsigned long long ga =
        (unsigned long long)(size_t)(G + (size_t)rbase * NDIM + kbase);
    v4u g0 = { 1u,                                   // count=1, user descriptor
               ldsoff,                               // lds_addr (bytes)
               (unsigned)ga,                         // global_addr[31:0]
               (unsigned)((ga >> 32) & 0x01ffffffu)  // global_addr[56:32]
                   | (2u << 30) };                   // type=2 ("image")
    v8i_ g1 = { (int)((1u << 16)    // data_size = 2 bytes
                    | (1u << 20)    // pad_enable
                    | (3u << 22)    // pad_interval: 16 DWORDs (=64B row)
                    | (3u << 25)),  // pad_amount:   4 DWORDs (=16B pad)
                (int)(2048u << 16),       // tensor_dim0[15:0]=2048 (<<16)
                (int)(2048u << 16),       // tensor_dim0[31:16]=0 | tensor_dim1[15:0]=2048
                (int)(32u << 16),         // tensor_dim1 hi=0 | tile_dim0=32 (k)
                128,                      // tile_dim1=128 (rows), tile_dim2=0
                2048,                     // tensor_dim0_stride low32
                0, 0 };
    v4i_ z4 = { 0, 0, 0, 0 };
#if __has_include(<hip/amd_detail/amd_gfx1250_TDM.h>)
    v8i_ z8 = { 0, 0, 0, 0, 0, 0, 0, 0 };
    __builtin_amdgcn_tensor_load_to_lds(g0, g1, z4, z4, z8, 0);   // clang-23 form
#else
    __builtin_amdgcn_tensor_load_to_lds(g0, g1, z4, z4, 0);       // ROCm 7.2 form
#endif
}
#else
#define HAVE_TDM 0
#endif

// Fallback staging: pure bf16 copy (no conversion), b128 in / b128 out.
__device__ __forceinline__ void stage_copy(const unsigned short* __restrict__ G,
                                           unsigned short* __restrict__ S,
                                           int rbase, int kbase, int tid) {
    int r  = tid >> 1;
    int kc = (tid & 1) * 16;
    const uint4* p = (const uint4*)(G + (size_t)(rbase + r) * NDIM + kbase + kc);
    uint4 u0 = p[0], u1 = p[1];
    if (kbase + 32 < NDIM) __builtin_prefetch(p + 4, 0, 1);   // next K-tile
    *(uint4*)((char*)S + (size_t)(r * LDT + kc) * 2)     = u0;
    *(uint4*)((char*)S + (size_t)(r * LDT + kc + 8) * 2) = u1;
}

// 16 WMMAs for one K-step: wave-level 64x32 output region, dual operand pairs.
__device__ __forceinline__ void wave_mma(const unsigned short* As1, const unsigned short* As2,
                                         const unsigned short* Bs1, const unsigned short* Bs2,
                                         int wm, int wn, int half, int lr, v8f acc[4][2]) {
    FragBF a1[4], a2[4], b1[2], b2[2];
    #pragma unroll
    for (int i = 0; i < 4; ++i) {
        int m = wm * 64 + i * 16 + lr;
        // A 16x32 bf16: lane-half 0 -> K0..7/K16..23, half 1 -> K8..15/K24..31
        a1[i].q[0] = lds_q(As1, m * LDT + half * 8);
        a1[i].q[1] = lds_q(As1, m * LDT + 16 + half * 8);
        a2[i].q[0] = lds_q(As2, m * LDT + half * 8);
        a2[i].q[1] = lds_q(As2, m * LDT + 16 + half * 8);
    }
    #pragma unroll
    for (int j = 0; j < 2; ++j) {
        int n = wn * 32 + j * 16 + lr;
        // B 32x16 bf16: lane-half 0 -> K0..7/K16..23, half 1 -> K8..15/K24..31
        b1[j].q[0] = lds_q(Bs1, n * LDT + half * 8);
        b1[j].q[1] = lds_q(Bs1, n * LDT + 16 + half * 8);
        b2[j].q[0] = lds_q(Bs2, n * LDT + half * 8);
        b2[j].q[1] = lds_q(Bs2, n * LDT + 16 + half * 8);
    }
    #pragma unroll
    for (int i = 0; i < 4; ++i)
        #pragma unroll
        for (int j = 0; j < 2; ++j) {
            acc[i][j] = __builtin_amdgcn_wmma_f32_16x16x32_bf16(
                false, a1[i].v, false, b1[j].v, (short)0, acc[i][j], false, false);
            acc[i][j] = __builtin_amdgcn_wmma_f32_16x16x32_bf16(
                false, a2[i].v, false, b2[j].v, (short)0, acc[i][j], false, false);
        }
}

// ---------------------------------------------------------------------------
// Fused dual GEMM:  Out = lrelu( A1 x B1^T + A2 x B2^T + bias )
// All operands bf16 [row][k] (ld=2048): A rows are M, B rows are N.
// BIASM: bias indexed by m (row) else by n (col). 128x128 block, BK=32,
// 8 waves (2 M x 4 N). TDM double-buffered staging when available.
// ---------------------------------------------------------------------------
template <bool BIASM>
__global__ __launch_bounds__(256)
void gemm_bf16_kernel(const unsigned short* __restrict__ A1, const unsigned short* __restrict__ A2,
                      const unsigned short* __restrict__ B1, const unsigned short* __restrict__ B2,
                      const float* __restrict__ bias, float* __restrict__ Out,
                      unsigned short* __restrict__ OutBf) {
    __shared__ __align__(16) unsigned short smem[2 * 4 * TILE_US];   // 80 KB

    const int tid  = threadIdx.x;
    const int mblk = blockIdx.y * 128;
    const int nblk = blockIdx.x * 128;
    const int wave = tid >> 5;
    const int lane = tid & 31;
    const int wm   = wave & 1;
    const int wn   = wave >> 1;
    const int half = lane >> 4;
    const int lr   = lane & 15;

    v8f acc[4][2];
    #pragma unroll
    for (int i = 0; i < 4; ++i)
        #pragma unroll
        for (int j = 0; j < 2; ++j) acc[i][j] = (v8f)0.0f;

#if HAVE_TDM
    const unsigned lds_base = (unsigned)(size_t)(const void*)smem;
    if (tid < 32) {   // wave 0 drives the Tensor Data Mover
        tdm_load_tile(A1, mblk, 0, lds_base + 0 * TILE_US * 2);
        tdm_load_tile(A2, mblk, 0, lds_base + 1 * TILE_US * 2);
        tdm_load_tile(B1, nblk, 0, lds_base + 2 * TILE_US * 2);
        tdm_load_tile(B2, nblk, 0, lds_base + 3 * TILE_US * 2);
    }
    for (int it = 0; it < NDIM / 32; ++it) {
        __syncthreads();              // next-buffer consumers from it-1 are done
        if (tid < 32) {
            if (it + 1 < NDIM / 32) {
                const unsigned nb = lds_base + (unsigned)(((it + 1) & 1) * 4) * TILE_US * 2;
                const int kn = (it + 1) * 32;
                tdm_load_tile(A1, mblk, kn, nb + 0 * TILE_US * 2);
                tdm_load_tile(A2, mblk, kn, nb + 1 * TILE_US * 2);
                tdm_load_tile(B1, nblk, kn, nb + 2 * TILE_US * 2);
                tdm_load_tile(B2, nblk, kn, nb + 3 * TILE_US * 2);
                __builtin_amdgcn_s_wait_tensorcnt(4);  // current buffer complete
            } else {
                __builtin_amdgcn_s_wait_tensorcnt(0);
            }
        }
        __syncthreads();              // current buffer published to all waves
        const unsigned short* base = smem + (size_t)((it & 1) * 4) * TILE_US;
        wave_mma(base, base + TILE_US, base + 2 * TILE_US, base + 3 * TILE_US,
                 wm, wn, half, lr, acc);
    }
#else
    for (int kt = 0; kt < NDIM; kt += 32) {
        __syncthreads();
        stage_copy(A1, smem + 0 * TILE_US, mblk, kt, tid);
        stage_copy(A2, smem + 1 * TILE_US, mblk, kt, tid);
        stage_copy(B1, smem + 2 * TILE_US, nblk, kt, tid);
        stage_copy(B2, smem + 3 * TILE_US, nblk, kt, tid);
        __syncthreads();
        wave_mma(smem, smem + TILE_US, smem + 2 * TILE_US, smem + 3 * TILE_US,
                 wm, wn, half, lr, acc);
    }
#endif

    // Epilogue: C/D layout -> n = lane%16, m = vgpr + 8*(lane/16)
    #pragma unroll
    for (int i = 0; i < 4; ++i)
        #pragma unroll
        for (int j = 0; j < 2; ++j) {
            int gm0 = mblk + wm * 64 + i * 16 + half * 8;
            int gn  = nblk + wn * 32 + j * 16 + lr;
            float bn = BIASM ? 0.0f : bias[gn];
            #pragma unroll
            for (int r = 0; r < 8; ++r) {
                float v = acc[i][j][r] + (BIASM ? bias[gm0 + r] : bn);
                v = (v > 0.0f) ? v : NEGSL * v;
                Out[(size_t)(gm0 + r) * NDIM + gn] = v;
                if (OutBf) OutBf[(size_t)(gm0 + r) * NDIM + gn] = f32_bf16(v);
            }
        }
}

// ---------------------------------------------------------------------------
// Support kernels
// ---------------------------------------------------------------------------

// hT = h^T (f32) and bf16 copy of hT.
__global__ __launch_bounds__(256)
void transpose_kernel(const float* __restrict__ in, float* __restrict__ out,
                      unsigned short* __restrict__ outbf) {
    __shared__ float t[32][33];
    int bx = blockIdx.x * 32, by = blockIdx.y * 32;
    int x = threadIdx.x, y = threadIdx.y;       // 32 x 8
    #pragma unroll
    for (int i = 0; i < 32; i += 8)
        t[y + i][x] = in[(size_t)(by + y + i) * NDIM + bx + x];
    __syncthreads();
    #pragma unroll
    for (int i = 0; i < 32; i += 8) {
        float v = t[x][y + i];
        size_t o = (size_t)(bx + y + i) * NDIM + by + x;
        out[o]   = v;
        outbf[o] = f32_bf16(v);
    }
}

// Weight prep: out_bf[c][r] = bf16(in[r][c])  (transpose + convert, once/launch)
__global__ __launch_bounds__(256)
void wprep_kernel(const float* __restrict__ in, unsigned short* __restrict__ out) {
    __shared__ float t[32][33];
    int bx = blockIdx.x * 32, by = blockIdx.y * 32;
    int x = threadIdx.x, y = threadIdx.y;
    #pragma unroll
    for (int i = 0; i < 32; i += 8)
        t[y + i][x] = in[(size_t)(by + y + i) * NDIM + bx + x];
    __syncthreads();
    #pragma unroll
    for (int i = 0; i < 32; i += 8)
        out[(size_t)(bx + y + i) * NDIM + by + x] = f32_bf16(t[x][y + i]);
}

__global__ __launch_bounds__(256)
void count_kernel(const int* __restrict__ tgt, float* __restrict__ cnt, int E) {
    int e = blockIdx.x * 256 + threadIdx.x;
    if (e < E) atomicAdd(&cnt[tgt[e]], 1.0f);
}

// acc[tgt[e]][c] += H[src[e]][c]   (rows contiguous -> coalesced atomics)
__global__ __launch_bounds__(256)
void aggregate_kernel(const float* __restrict__ H, const int* __restrict__ src,
                      const int* __restrict__ tgt, float* __restrict__ acc) {
    int e = blockIdx.x;
    int c = blockIdx.y * 256 + threadIdx.x;
    int s = src[e], t = tgt[e];
    atomicAdd(&acc[(size_t)t * NDIM + c], H[(size_t)s * NDIM + c]);
}

// mean_bf = bf16(acc / max(cnt_row, 1))
__global__ __launch_bounds__(256)
void scale_bf_kernel(const float* __restrict__ acc, const float* __restrict__ cnt,
                     unsigned short* __restrict__ outbf) {
    int idx = blockIdx.x * 256 + threadIdx.x;
    int r = idx >> 11;
    outbf[idx] = f32_bf16(acc[idx] / fmaxf(cnt[r], 1.0f));
}

// ---------------------------------------------------------------------------
// Host orchestration
// ---------------------------------------------------------------------------
extern "C" void kernel_launch(void* const* d_in, const int* in_sizes, int n_in,
                              void* d_out, int out_size, void* d_ws, size_t ws_size,
                              hipStream_t stream) {
    const float* x      = (const float*)d_in[0];
    const int*   knn    = (const int*)d_in[1];
    const int*   ppi    = (const int*)d_in[2];
    const float* Wsrc[4] = { (const float*)d_in[3], (const float*)d_in[5],
                             (const float*)d_in[6], (const float*)d_in[8] };
    const float* col_bl = (const float*)d_in[4];
    const float* row_bl = (const float*)d_in[7];
    float* out = (float*)d_out;

    const int* knn_src = knn;  const int* knn_tgt = knn + EKNN;
    const int* ppi_src = ppi;  const int* ppi_tgt = ppi + EPPI;

    const size_t MAT = (size_t)NDIM * NDIM;            // 4M elems
    char* w = (char*)d_ws;
    float*          acc     = (float*)(w);                       //  0 .. 16MB
    float*          hT      = (float*)(w + MAT * 4);             // 16 .. 32MB
    float*          hMid    = (float*)(w + MAT * 8);             // 32 .. 48MB
    float*          hRow    = (float*)(w + MAT * 12);            // 48 .. 64MB
    unsigned short* hT_bf   = (unsigned short*)(w + MAT * 16);   // 64 .. 72MB
    unsigned short* mean_bf = (unsigned short*)(w + MAT * 18);   // 72 .. 80MB
    unsigned short* hMid_bf = (unsigned short*)(w + MAT * 20);   // 80 .. 88MB
    float*          cntC    = (float*)(w + MAT * 22);            // 88MB
    float*          cntG    = cntC + NDIM;
    unsigned short* wbf     = (unsigned short*)(w + MAT * 24);   // 96 .. 224MB
    // wbf layout: [tensor 0..3][layer 0..3], each 2048x2048 bf16 (transposed)
    //   tensor 0: col_Wl^T   1: col_Wr^T   2: row_Wl^T   3: row_Wr^T

    const dim3 gT(64, 64), bT(32, 8), gG(16, 16);

    // ---- one-time prep: counts + transposed bf16 weights
    hipMemsetAsync(cntC, 0, 2 * NDIM * sizeof(float), stream);
    count_kernel<<<(EKNN + 255) / 256, 256, 0, stream>>>(knn_tgt, cntC, EKNN);
    count_kernel<<<(EPPI + 255) / 256, 256, 0, stream>>>(ppi_tgt, cntG, EPPI);
    for (int t = 0; t < 4; ++t)
        for (int l = 0; l < LAYERS; ++l)
            wprep_kernel<<<gT, bT, 0, stream>>>(Wsrc[t] + (size_t)l * MAT,
                                                wbf + ((size_t)t * LAYERS + l) * MAT);

    const float* hIn = x;
    for (int l = 0; l < LAYERS; ++l) {
        const size_t bo = (size_t)l * NDIM;
        const unsigned short* cWlT = wbf + ((size_t)0 * LAYERS + l) * MAT;
        const unsigned short* cWrT = wbf + ((size_t)1 * LAYERS + l) * MAT;
        const unsigned short* rWlT = wbf + ((size_t)2 * LAYERS + l) * MAT;
        const unsigned short* rWrT = wbf + ((size_t)3 * LAYERS + l) * MAT;

        // ---- cols side (cells/KNN): out = lrelu(Wl^T meanC + Wr^T h + bl[g])
        transpose_kernel<<<gT, bT, 0, stream>>>(hIn, hT, hT_bf);
        hipMemsetAsync(acc, 0, MAT * sizeof(float), stream);
        aggregate_kernel<<<dim3(EKNN, NDIM / 256), 256, 0, stream>>>(hT, knn_src, knn_tgt, acc);
        scale_bf_kernel<<<(int)(MAT / 256), 256, 0, stream>>>(acc, cntC, mean_bf);
        gemm_bf16_kernel<true><<<gG, 256, 0, stream>>>(
            cWlT, cWrT, mean_bf, hT_bf, col_bl + bo, hMid, hMid_bf);

        // ---- rows side (genes/PPI): out = lrelu(meanG Wl + h Wr + bl[c])
        hipMemsetAsync(acc, 0, MAT * sizeof(float), stream);
        aggregate_kernel<<<dim3(EPPI, NDIM / 256), 256, 0, stream>>>(hMid, ppi_src, ppi_tgt, acc);
        scale_bf_kernel<<<(int)(MAT / 256), 256, 0, stream>>>(acc, cntG, mean_bf);
        float* outL = (l == LAYERS - 1) ? out : hRow;
        gemm_bf16_kernel<false><<<gG, 256, 0, stream>>>(
            mean_bf, hMid_bf, rWlT, rWrT, row_bl + bo, outL, (unsigned short*)nullptr);
        hIn = outL;
    }
    (void)in_sizes; (void)n_in; (void)out_size; (void)ws_size;
}
